// MultiHeadAttention_67765993996452
// MI455X (gfx1250) — matrix operand order repaired and checked by reading, not compiled, not run on hardware
//
#include <hip/hip_runtime.h>
#include <hip/hip_bf16.h>
#include <math.h>

// MHA forward for MI455X (gfx1250, wave32, WMMA).
// B=4, S=2048, D_MODEL=1024, H=16, D_K=64.

#define B_  4
#define S_  2048
#define DM_ 1024
#define H_  16
#define DK_ 64

typedef __bf16 bf16;
typedef __attribute__((ext_vector_type(16))) __bf16 v16bf;
typedef __attribute__((ext_vector_type(8)))  __bf16 v8bf;
typedef __attribute__((ext_vector_type(8)))  float  v8f;
typedef __attribute__((ext_vector_type(4)))  float  f32x4;

static __device__ __forceinline__ v8f vzero8() {
  v8f z = {0.f, 0.f, 0.f, 0.f, 0.f, 0.f, 0.f, 0.f};
  return z;
}

static __device__ __forceinline__ v8f wmma_bf16(v16bf a, v16bf b, v8f c) {
  // D = A(16x32 bf16) * B(32x16 bf16) + C(16x16 f32)
  return __builtin_amdgcn_wmma_f32_16x16x32_bf16(
      /*neg_a=*/false, a, /*neg_b=*/false, b,
      /*c_mod=*/(short)0, c, /*reuse_a=*/false, /*reuse_b=*/false);
}

// ---- A fragment (16x32 bf16): lane holds row m; lanes 0-15: K=0..15,
// lanes 16-31: K=16..31 (contiguous, matching B).
static __device__ __forceinline__ v16bf afrag_bf16(const bf16* __restrict__ row,
                                                   int kk, int lane) {
  int kb = (lane < 16) ? 0 : 16;
  v8bf lo = *(const v8bf*)(row + kk + kb);
  v8bf hi = *(const v8bf*)(row + kk + kb + 8);
  v16bf a;
#pragma unroll
  for (int i = 0; i < 8; ++i) {
    a[i]     = lo[i];
    a[i + 8] = hi[i];
  }
  return a;
}

// ---- B fragment (32x16 bf16): lane holds col n; lanes 0-15: K=0..15,
// lanes 16-31: K=16..31; K contiguous along the source row.
static __device__ __forceinline__ v16bf bfrag_bf16(const bf16* __restrict__ row,
                                                   int kk, int lane) {
  int kb = (lane < 16) ? 0 : 16;
  return *(const v16bf*)(row + kk + kb);
}

// ============================================================================
// Kernel 0: elementwise f32 -> bf16 staging (bandwidth bound).
// ============================================================================
__global__ __launch_bounds__(256) void cvt_f32_bf16_kernel(
    const float* __restrict__ in, bf16* __restrict__ out, int n8) {
  int i = blockIdx.x * 256 + threadIdx.x;
  if (i < n8) {
    const f32x4* p = (const f32x4*)(in + (size_t)i * 8);
    f32x4 x0 = p[0], x1 = p[1];
    v8bf o;
#pragma unroll
    for (int j = 0; j < 4; ++j) {
      o[j]     = (__bf16)x0[j];
      o[j + 4] = (__bf16)x1[j];
    }
    *(v8bf*)(out + (size_t)i * 8) = o;
  }
}

// ============================================================================
// Kernel 1: projection  Y = X @ W^T + b  (X: bf16 [8192,1024], W: bf16
// [1024,1024] row-major (out,in)).  Output bf16:
//   transposeOut==0 : dst[b,h,s,dk]  (Q and K)
//   transposeOut==1 : dst[b,h,dk,s]  (V transposed for the P*V WMMA)
// Block = 256 threads = 8 waves (2 row-groups x 4 col-groups).
// Per-wave tile: 32 x 64  (8 WMMAs per k-step from 2 A-frags + 4 B-frags).
// Block tile: 64 x 256.
// ============================================================================
__global__ __launch_bounds__(256) void proj_qkv_kernel(
    const bf16* __restrict__ X, const bf16* __restrict__ W,
    const float* __restrict__ bias, bf16* __restrict__ dst, int transposeOut) {
  const int lane = threadIdx.x & 31;
  const int wid  = threadIdx.x >> 5;
  const int ln   = lane & 15;
  const int half = lane >> 4;
  const int mBase = blockIdx.y * 64 + (wid >> 2) * 32;
  const int nBase = blockIdx.x * 256 + (wid & 3) * 64;

  v8f acc[2][4];
#pragma unroll
  for (int u = 0; u < 2; ++u)
#pragma unroll
    for (int t = 0; t < 4; ++t) acc[u][t] = vzero8();

  const bf16* xrow0 = X + (size_t)(mBase + ln) * DM_;
  const bf16* xrow1 = xrow0 + (size_t)16 * DM_;
  for (int kk = 0; kk < DM_; kk += 32) {
    v16bf a0 = afrag_bf16(xrow0, kk, lane);
    v16bf a1 = afrag_bf16(xrow1, kk, lane);
#pragma unroll
    for (int t = 0; t < 4; ++t) {
      int n = nBase + t * 16 + ln;
      v16bf b = bfrag_bf16(W + (size_t)n * DM_, kk, lane);
      acc[0][t] = wmma_bf16(a0, b, acc[0][t]);
      acc[1][t] = wmma_bf16(a1, b, acc[1][t]);
    }
  }

  // Epilogue: bias + scatter into head-major layout.
#pragma unroll
  for (int u = 0; u < 2; ++u) {
    int mB = mBase + u * 16;
#pragma unroll
    for (int t = 0; t < 4; ++t) {
      int e  = nBase + t * 16 + ln;   // output feature
      int h  = e >> 6;
      int dk = e & 63;
      float bv = bias[e];
      if (transposeOut == 0) {
#pragma unroll
        for (int r = 0; r < 8; ++r) {
          int m  = mB + r + 8 * half;        // token index b*S+s
          int bb = m >> 11;
          int s  = m & (S_ - 1);
          dst[(((size_t)(bb * H_ + h)) * S_ + s) * DK_ + dk] =
              (__bf16)(acc[u][t][r] + bv);
        }
      } else {
        int m0 = mB + 8 * half;              // 8 consecutive tokens
        int bb = m0 >> 11;
        int s0 = m0 & (S_ - 1);
        v8bf ov;
#pragma unroll
        for (int r = 0; r < 8; ++r) ov[r] = (__bf16)(acc[u][t][r] + bv);
        *(v8bf*)(dst + (((size_t)(bb * H_ + h)) * DK_ + dk) * S_ + s0) = ov;
      }
    }
  }
}

// ============================================================================
// Kernel 2: flash attention per (b,h).  Each wave owns 16 query rows; keys are
// processed in chunks of 32 with online softmax.  Q/K: bf16 [BH, S, 64];
// Vt: bf16 [BH, 64, S]; output attn: bf16 [B*S, 1024] (head-interleaved).
// ============================================================================
__global__ __launch_bounds__(256) void flash_attn_kernel(
    const bf16* __restrict__ Q, const bf16* __restrict__ K,
    const bf16* __restrict__ Vt, bf16* __restrict__ O) {
  __shared__ __bf16 Plds[8][16 * 32];   // per-wave 16x32 P tile (bf16)

  const int lane = threadIdx.x & 31;
  const int wid  = threadIdx.x >> 5;
  const int ln   = lane & 15;
  const int half = lane >> 4;
  const int bh    = blockIdx.y;                  // 0..63
  const int qBase = blockIdx.x * 128 + wid * 16; // 16 q rows per wave

  const bf16* Qb = Q  + (size_t)bh * S_ * DK_;
  const bf16* Kb = K  + (size_t)bh * S_ * DK_;
  const bf16* Vb = Vt + (size_t)bh * DK_ * S_;

  // Hold the 16x64 Q tile in registers as two A fragments (d 0..31, 32..63).
  const bf16* qrow = Qb + (size_t)(qBase + ln) * DK_;
  v16bf aq0 = afrag_bf16(qrow, 0, lane);
  v16bf aq1 = afrag_bf16(qrow, 32, lane);

  v8f o[4];
#pragma unroll
  for (int t = 0; t < 4; ++t) o[t] = vzero8();
  float mrun[8], lrun[8];
#pragma unroll
  for (int r = 0; r < 8; ++r) { mrun[r] = -3.0e38f; lrun[r] = 0.f; }

  for (int kc = 0; kc < S_; kc += 32) {
    // ---- scores: 16 q-rows x 32 keys, accumulated over d_k = 64
    v8f s0 = vzero8(), s1 = vzero8();
    {
      const bf16* k0 = Kb + (size_t)(kc + ln) * DK_;
      const bf16* k1 = Kb + (size_t)(kc + 16 + ln) * DK_;
      s0 = wmma_bf16(aq0, bfrag_bf16(k0, 0, lane), s0);
      s0 = wmma_bf16(aq1, bfrag_bf16(k0, 32, lane), s0);
      s1 = wmma_bf16(aq0, bfrag_bf16(k1, 0, lane), s1);
      s1 = wmma_bf16(aq1, bfrag_bf16(k1, 32, lane), s1);
    }

    // ---- online softmax (row reductions across the 16-lane C-tile groups)
#pragma unroll
    for (int r = 0; r < 8; ++r) {
      float x0 = s0[r] * 0.125f;   // 1/sqrt(64)
      float x1 = s1[r] * 0.125f;
      float cm = fmaxf(x0, x1);
      cm = fmaxf(cm, __shfl_xor(cm, 1, 32));
      cm = fmaxf(cm, __shfl_xor(cm, 2, 32));
      cm = fmaxf(cm, __shfl_xor(cm, 4, 32));
      cm = fmaxf(cm, __shfl_xor(cm, 8, 32));
      float nm   = fmaxf(mrun[r], cm);
      float corr = __expf(mrun[r] - nm);
      float p0   = __expf(x0 - nm);
      float p1   = __expf(x1 - nm);
      float rs = p0 + p1;
      rs += __shfl_xor(rs, 1, 32);
      rs += __shfl_xor(rs, 2, 32);
      rs += __shfl_xor(rs, 4, 32);
      rs += __shfl_xor(rs, 8, 32);
      lrun[r] = lrun[r] * corr + rs;
      mrun[r] = nm;
      o[0][r] *= corr; o[1][r] *= corr; o[2][r] *= corr; o[3][r] *= corr;
      int mrow = r + 8 * half;
      Plds[wid][mrow * 32 + ln]      = (__bf16)p0;
      Plds[wid][mrow * 32 + 16 + ln] = (__bf16)p1;
    }

    // ---- P (16x32) * V-chunk (32x64): same-wave LDS roundtrip (in-order DS)
    v16bf ap = afrag_bf16(&Plds[wid][(size_t)ln * 32], 0, lane);
#pragma unroll
    for (int t = 0; t < 4; ++t) {
      v16bf bv = bfrag_bf16(Vb + (size_t)(t * 16 + ln) * S_ + kc, 0, lane);
      o[t] = wmma_bf16(ap, bv, o[t]);
    }
  }

  // ---- epilogue: normalize and write head-interleaved bf16 [B*S, 1024]
  const int bb = bh >> 4;
  const int h  = bh & 15;
#pragma unroll
  for (int r = 0; r < 8; ++r) {
    float inv = 1.0f / lrun[r];
    int s = qBase + r + 8 * half;
#pragma unroll
    for (int t = 0; t < 4; ++t) {
      O[((size_t)(bb * S_ + s)) * DM_ + h * DK_ + t * 16 + ln] =
          (__bf16)(o[t][r] * inv);
    }
  }
}

// ============================================================================
// Kernel 3: output projection  out = attn @ W_o^T + b_o   (f32 output)
// Same 32x64-per-wave tiling as the QKV projections.
// ============================================================================
__global__ __launch_bounds__(256) void out_proj_kernel(
    const bf16* __restrict__ A, const bf16* __restrict__ W,
    const float* __restrict__ bias, float* __restrict__ Y) {
  const int lane = threadIdx.x & 31;
  const int wid  = threadIdx.x >> 5;
  const int ln   = lane & 15;
  const int half = lane >> 4;
  const int mBase = blockIdx.y * 64 + (wid >> 2) * 32;
  const int nBase = blockIdx.x * 256 + (wid & 3) * 64;

  v8f acc[2][4];
#pragma unroll
  for (int u = 0; u < 2; ++u)
#pragma unroll
    for (int t = 0; t < 4; ++t) acc[u][t] = vzero8();

  const bf16* arow0 = A + (size_t)(mBase + ln) * DM_;
  const bf16* arow1 = arow0 + (size_t)16 * DM_;
  for (int kk = 0; kk < DM_; kk += 32) {
    v16bf a0 = afrag_bf16(arow0, kk, lane);
    v16bf a1 = afrag_bf16(arow1, kk, lane);
#pragma unroll
    for (int t = 0; t < 4; ++t) {
      int n = nBase + t * 16 + ln;
      v16bf b = bfrag_bf16(W + (size_t)n * DM_, kk, lane);
      acc[0][t] = wmma_bf16(a0, b, acc[0][t]);
      acc[1][t] = wmma_bf16(a1, b, acc[1][t]);
    }
  }

#pragma unroll
  for (int u = 0; u < 2; ++u) {
    int mB = mBase + u * 16;
#pragma unroll
    for (int t = 0; t < 4; ++t) {
      int e = nBase + t * 16 + ln;
      float bv = bias[e];
#pragma unroll
      for (int r = 0; r < 8; ++r) {
        int m = mB + r + 8 * half;
        Y[(size_t)m * DM_ + e] = acc[u][t][r] + bv;
      }
    }
  }
}

// ============================================================================
extern "C" void kernel_launch(void* const* d_in, const int* in_sizes, int n_in,
                              void* d_out, int out_size, void* d_ws,
                              size_t ws_size, hipStream_t stream) {
  (void)in_sizes; (void)n_in; (void)out_size; (void)ws_size;

  const float* q  = (const float*)d_in[0];
  const float* k  = (const float*)d_in[1];
  const float* v  = (const float*)d_in[2];
  const float* wq = (const float*)d_in[3];
  const float* bq = (const float*)d_in[4];
  const float* wk = (const float*)d_in[5];
  const float* bk = (const float*)d_in[6];
  const float* wv = (const float*)d_in[7];
  const float* bv = (const float*)d_in[8];
  const float* wo = (const float*)d_in[9];
  const float* bo = (const float*)d_in[10];
  float* out = (float*)d_out;

  const size_t XE = (size_t)B_ * S_ * DM_;          // 8,388,608 elems
  const size_t WE = (size_t)DM_ * DM_;              // 1,048,576 elems

  // Workspace layout (bf16 elems):
  //   Xq, Xk, Xv : staged activations     (3 * 16 MB)
  //   Wqb..Wob   : staged weights         (4 *  2 MB)
  //   Qb, Kb, Vt : projected heads        (3 * 16 MB)
  //   AO         : attention out, aliases Xq (dead by then)
  bf16* Xq  = (bf16*)d_ws;
  bf16* Xk  = Xq + XE;
  bf16* Xv  = Xk + XE;
  bf16* Wqb = Xv + XE;
  bf16* Wkb = Wqb + WE;
  bf16* Wvb = Wkb + WE;
  bf16* Wob = Wvb + WE;
  bf16* Qb  = Wob + WE;
  bf16* Kb  = Qb + XE;
  bf16* Vt  = Kb + XE;   // [BH, 64, S] transposed
  bf16* AO  = Xq;        // alias: attn output bf16 [B*S, 1024]

  dim3 blk(256);
  dim3 gCvtX((int)(XE / 8 + 255) / 256);
  dim3 gCvtW((int)(WE / 8 + 255) / 256);
  dim3 gProj(DM_ / 256, (B_ * S_) / 64);   // (4, 128)
  dim3 gAttn(S_ / 128, B_ * H_);           // (16, 64)

  cvt_f32_bf16_kernel<<<gCvtX, blk, 0, stream>>>(q, Xq, (int)(XE / 8));
  cvt_f32_bf16_kernel<<<gCvtX, blk, 0, stream>>>(k, Xk, (int)(XE / 8));
  cvt_f32_bf16_kernel<<<gCvtX, blk, 0, stream>>>(v, Xv, (int)(XE / 8));
  cvt_f32_bf16_kernel<<<gCvtW, blk, 0, stream>>>(wq, Wqb, (int)(WE / 8));
  cvt_f32_bf16_kernel<<<gCvtW, blk, 0, stream>>>(wk, Wkb, (int)(WE / 8));
  cvt_f32_bf16_kernel<<<gCvtW, blk, 0, stream>>>(wv, Wvb, (int)(WE / 8));
  cvt_f32_bf16_kernel<<<gCvtW, blk, 0, stream>>>(wo, Wob, (int)(WE / 8));

  proj_qkv_kernel<<<gProj, blk, 0, stream>>>(Xq, Wqb, bq, Qb, 0);
  proj_qkv_kernel<<<gProj, blk, 0, stream>>>(Xk, Wkb, bk, Kb, 0);
  proj_qkv_kernel<<<gProj, blk, 0, stream>>>(Xv, Wvb, bv, Vt, 1);
  flash_attn_kernel<<<gAttn, blk, 0, stream>>>(Qb, Kb, Vt, AO);
  out_proj_kernel<<<gProj, blk, 0, stream>>>(AO, Wob, bo, out);
}
